// HGNN_64029372449468
// MI455X (gfx1250) — matrix operand order repaired and checked by reading, not compiled, not run on hardware
//
#include <hip/hip_runtime.h>
#include <hip/hip_bf16.h>

// ---------------------------------------------------------------------------
// HGNN link prediction for MI455X (gfx1250, wave32, WMMA)
//
// Pipeline:
//   1. degrees dv/de via f32 atomics; invert in place (dv_isqrt, de_inv)
//   2. bf16 conversion of X, W1, W2
//   3. GEMM1 (WMMA bf16->f32):  P = (X@W1 + b1) * dv_isqrt          [N,256]
//   4. seg-scan scatter (sorted inc_e):  Q[e] += P[v]               [E,256]
//   5. atomic scatter:  R[v] += Q[e]*de_inv                          [N,256]
//   6. H = bf16(relu(R * dv_isqrt))  (into P region)                 [N,256]
//   7. GEMM2 (WMMA):  Q = (H@W2 + b2) * dv_isqrt                     [N,128]
//   8. seg-scan scatter:  R[e] += Q[v]                               [E,128]
//   9. atomic scatter:  P[v] += R[e]*de_inv                          [N,128]
//  10. z = relu(P * dv_isqrt)  (into Q)                              [N,128]
//  11. link pooling: sigmoid(0.5*(z[a]+z[b]) . fcW + fcb)            [L,1]
// ---------------------------------------------------------------------------

typedef __bf16 bf16_t;
typedef __attribute__((ext_vector_type(16))) __bf16 v16bf;
typedef __attribute__((ext_vector_type(8)))  __bf16 v8bf;
typedef __attribute__((ext_vector_type(8)))  float  v8f;

#define HG_N_NODES  100000
#define HG_N_HEDGES 100000
#define HG_N_INC    3200000
#define HG_N_LINKS  262144

// -------------------------- degree / scaling kernels -----------------------

__global__ void hg_degree_kernel(const int* __restrict__ inc_v,
                                 const int* __restrict__ inc_e,
                                 float* dv, float* de, int n) {
  int i = blockIdx.x * blockDim.x + threadIdx.x;
  if (i < n) {
    atomicAdd(dv + inc_v[i], 1.0f);
    atomicAdd(de + inc_e[i], 1.0f);
  }
}

__global__ void hg_invdeg_kernel(float* dv, float* de, int n) {
  int i = blockIdx.x * blockDim.x + threadIdx.x;
  if (i < n) {
    float a = dv[i];
    dv[i] = (a > 0.0f) ? rsqrtf(fmaxf(a, 1.0f)) : 0.0f;
    float b = de[i];
    de[i] = (b > 0.0f) ? (1.0f / fmaxf(b, 1.0f)) : 0.0f;
  }
}

__global__ void hg_cvt_bf16_kernel(const float* __restrict__ in,
                                   bf16_t* __restrict__ out, int n) {
  int i = blockIdx.x * blockDim.x + threadIdx.x;
  if (i < n) out[i] = (bf16_t)in[i];
}

// relu(in * rowScale[row]) -> bf16   (row = i >> logC)
__global__ void hg_relu_scale_cvt_kernel(const float* __restrict__ in,
                                         const float* __restrict__ rowScale,
                                         bf16_t* __restrict__ out,
                                         int n, int logC) {
  int i = blockIdx.x * blockDim.x + threadIdx.x;
  if (i < n) {
    float v = in[i] * rowScale[i >> logC];
    out[i] = (bf16_t)fmaxf(v, 0.0f);
  }
}

// relu(in * rowScale[row]) -> f32
__global__ void hg_relu_scale_kernel(const float* __restrict__ in,
                                     const float* __restrict__ rowScale,
                                     float* __restrict__ out,
                                     int n, int logC) {
  int i = blockIdx.x * blockDim.x + threadIdx.x;
  if (i < n) {
    float v = in[i] * rowScale[i >> logC];
    out[i] = fmaxf(v, 0.0f);
  }
}

// ----------------- atomic scatter-add (unsorted destination) ---------------
// dst[dIdx[inc], c..c+3] += src[sIdx[inc], c..c+3] * (sScale ? sScale[sIdx]:1)
// One thread per (incidence, 4-channel chunk); reads coalesced along the row.
__global__ void hg_scatter_add_kernel(const float* __restrict__ src,
                                      float* __restrict__ dst,
                                      const int* __restrict__ sIdx,
                                      const int* __restrict__ dIdx,
                                      const float* __restrict__ sScale,
                                      int C4, int nInc) {
  long long t = (long long)blockIdx.x * blockDim.x + threadIdx.x;
  int inc = (int)(t / C4);
  if (inc >= nInc) return;
  int c = ((int)(t % C4)) * 4;
  int s = sIdx[inc];
  int d = dIdx[inc];
  const float4 v = *(const float4*)(src + (size_t)s * (C4 * 4) + c);
  float sc = sScale ? sScale[s] : 1.0f;
  float* dp = dst + (size_t)d * (C4 * 4) + c;
  atomicAdd(dp + 0, v.x * sc);
  atomicAdd(dp + 1, v.y * sc);
  atomicAdd(dp + 2, v.z * sc);
  atomicAdd(dp + 3, v.w * sc);
}

// ----------- segmented-scan scatter-add (sorted destination rows) ----------
// dst[eIdx[inc], c..c+3] += src[vIdx[inc], c..c+3]
// inc_e is sorted: a wave32 handles 32 consecutive incidences for one
// 4-channel chunk, combines equal-destination runs via a segmented inclusive
// scan in registers, and only the last lane of each run issues atomics.
__global__ void hg_scatter_seg_kernel(const float* __restrict__ src,
                                      float* __restrict__ dst,
                                      const int* __restrict__ vIdx,
                                      const int* __restrict__ eIdx,
                                      int C, int nInc) {
  int inc  = blockIdx.x * blockDim.x + threadIdx.x;
  int lane = threadIdx.x & 31;
  if (inc >= nInc) return;
  int c = blockIdx.y * 4;
  int v = vIdx[inc];
  int e = eIdx[inc];
  const float4 val = *(const float4*)(src + (size_t)v * C + c);
  float x0 = val.x, x1 = val.y, x2 = val.z, x3 = val.w;
#pragma unroll
  for (int d = 1; d < 32; d <<= 1) {
    int   eu = __shfl_up(e, d, 32);
    float y0 = __shfl_up(x0, d, 32);
    float y1 = __shfl_up(x1, d, 32);
    float y2 = __shfl_up(x2, d, 32);
    float y3 = __shfl_up(x3, d, 32);
    if (lane >= d && eu == e) { x0 += y0; x1 += y1; x2 += y2; x3 += y3; }
  }
  int en = __shfl_down(e, 1, 32);
  bool last = (lane == 31) || (en != e);
  if (last) {
    float* dp = dst + (size_t)e * C + c;
    atomicAdd(dp + 0, x0);
    atomicAdd(dp + 1, x1);
    atomicAdd(dp + 2, x2);
    atomicAdd(dp + 3, x3);
  }
}

// ------------------------------ WMMA GEMM ----------------------------------
// Out[row, col] = (sum_k A[row,k]*B[k,col] + bias[col]) * rowScale[row]
// A: [Nrows x K] bf16 row-major, B: [K x M] bf16 row-major.
// Block: 256 threads (8 wave32), tile 32 rows x 64 cols; wave w owns the
// 16x16 subtile (rsub = w>>2, csub = w&3). Full K staged in LDS (K <= 256).
// B is staged TRANSPOSED (BsT[col][k]) so each lane's fragment is contiguous
// -> ds_load_b128 instead of 16x ds_load_u16. LDS rows padded by 16 bf16
// (keeps 32B alignment, spreads banks).
__global__ __launch_bounds__(256) void hg_gemm_wmma_kernel(
    const bf16_t* __restrict__ A, const bf16_t* __restrict__ B,
    const float* __restrict__ bias, const float* __restrict__ rowScale,
    float* __restrict__ Out, int K, int M) {
  __shared__ bf16_t As[32 * (256 + 16)];   // ~17.4 KB
  __shared__ bf16_t BsT[64 * (256 + 16)];  // ~34.8 KB

  const int tid = threadIdx.x;
  const int rowBlk = blockIdx.x * 32;
  const int colBlk = blockIdx.y * 64;
  const int KP = K + 16;  // padded LDS row stride (32B-aligned)

  // Stage A tile: 32 rows x K (row-major, padded stride)
  for (int i = tid; i < 32 * K; i += 256) {
    int r = i / K, k = i % K;
    As[r * KP + k] = A[(size_t)(rowBlk + r) * K + k];
  }
  // Stage B tile transposed: BsT[c][k] = B[k][colBlk + c]
  // (global reads coalesced in (k,c) order; LDS writes strided but banked)
  for (int i = tid; i < K * 64; i += 256) {
    int k = i >> 6, c = i & 63;
    BsT[c * KP + k] = B[(size_t)k * M + colBlk + c];
  }
  __syncthreads();

  const int wave = tid >> 5;
  const int lane = tid & 31;
  const int hl   = lane >> 4;         // lane-half: 0 or 1
  const int l15  = lane & 15;
  const int rsub = (wave >> 2) * 16;  // 0 or 16
  const int csub = (wave & 3) * 16;   // 0,16,32,48

  const bf16_t* arow = &As[(rsub + l15) * KP];
  const bf16_t* bcol = &BsT[(csub + l15) * KP];

  v8f acc = {};
  for (int kk = 0; kk < K; kk += 32) {
    // A fragment (16x32): lanes 0-15: elems 0..15 = K 0..15;
    //                     lanes 16-31: elems 0..15 = K 16..31.
    const int khi = hl * 16;
    v8bf alo = *(const v8bf*)(arow + kk + khi);        // 16B aligned
    v8bf ahi = *(const v8bf*)(arow + kk + 8 + khi);    // 16B aligned
    v16bf a = __builtin_shufflevector(alo, ahi, 0, 1, 2, 3, 4, 5, 6, 7,
                                      8, 9, 10, 11, 12, 13, 14, 15);
    // B fragment (32x16): lane holds col (lane&15); elem i = K (kb+i),
    // kb = 0 for lanes 0-15, 16 for lanes 16-31.  Contiguous in BsT.
    const int kb = hl * 16;
    v16bf b = *(const v16bf*)(bcol + kk + kb);         // 32B aligned

    acc = __builtin_amdgcn_wmma_f32_16x16x32_bf16(
        false, a, false, b, (short)0, acc, false, false);
  }

  // C/D layout: elem j -> M = j + 8*hl, N = lane&15
  const int col = colBlk + csub + l15;
  const float bc = bias[col];
#pragma unroll
  for (int j = 0; j < 8; ++j) {
    int row = rowBlk + rsub + j + hl * 8;
    Out[(size_t)row * M + col] = (acc[j] + bc) * rowScale[row];
  }
}

// ----------------------------- link pooling --------------------------------
// One wave32 per link: pooled = 0.5*(z[a]+z[b]); out = sigmoid(pooled.fcW+fcb)
__global__ void hg_link_kernel(const float* __restrict__ z,
                               const int* __restrict__ link,
                               const float* __restrict__ fcW,
                               const float* __restrict__ fcb,
                               float* __restrict__ out, int nLinks) {
  int gt   = blockIdx.x * blockDim.x + threadIdx.x;
  int wid  = gt >> 5;
  int lane = gt & 31;
  if (wid >= nLinks) return;
  int a = link[(size_t)wid * 2 + 0];
  int b = link[(size_t)wid * 2 + 1];
  float4 pa = ((const float4*)(z + (size_t)a * 128))[lane];
  float4 pb = ((const float4*)(z + (size_t)b * 128))[lane];
  float4 w  = ((const float4*)fcW)[lane];
  float p = 0.5f * ((pa.x + pb.x) * w.x + (pa.y + pb.y) * w.y +
                    (pa.z + pb.z) * w.z + (pa.w + pb.w) * w.w);
#pragma unroll
  for (int m = 16; m >= 1; m >>= 1) p += __shfl_xor(p, m, 32);
  if (lane == 0) out[wid] = 1.0f / (1.0f + __expf(-(p + fcb[0])));
}

// ------------------------------- launcher ----------------------------------

extern "C" void kernel_launch(void* const* d_in, const int* in_sizes, int n_in,
                              void* d_out, int out_size, void* d_ws,
                              size_t ws_size, hipStream_t stream) {
  const float* X   = (const float*)d_in[0];
  const float* W1  = (const float*)d_in[1];
  const float* b1  = (const float*)d_in[2];
  const float* W2  = (const float*)d_in[3];
  const float* b2  = (const float*)d_in[4];
  const float* fcW = (const float*)d_in[5];
  const float* fcb = (const float*)d_in[6];
  const int* inc_v = (const int*)d_in[7];
  const int* inc_e = (const int*)d_in[8];
  const int* link  = (const int*)d_in[9];
  float* out = (float*)d_out;
  (void)in_sizes; (void)n_in; (void)out_size; (void)ws_size;

  char* ws = (char*)d_ws;
  size_t off = 0;
  auto carve = [&](size_t bytes) -> char* {
    char* p = ws + off;
    off += (bytes + 255) & ~(size_t)255;
    return p;
  };
  float*  dv  = (float*)carve((size_t)HG_N_NODES * 4);   // -> dv_isqrt
  float*  de  = (float*)carve((size_t)HG_N_HEDGES * 4);  // -> de_inv
  bf16_t* W1b = (bf16_t*)carve((size_t)128 * 256 * 2);
  bf16_t* W2b = (bf16_t*)carve((size_t)256 * 128 * 2);
  bf16_t* Xb  = (bf16_t*)carve((size_t)HG_N_NODES * 128 * 2);
  float*  P   = (float*)carve((size_t)HG_N_NODES * 256 * 4);
  float*  Q   = (float*)carve((size_t)HG_N_NODES * 256 * 4);
  float*  R   = (float*)carve((size_t)HG_N_NODES * 256 * 4);

  // 1. degrees + inverse scalings
  hipMemsetAsync(dv, 0, (size_t)HG_N_NODES * 4, stream);
  hipMemsetAsync(de, 0, (size_t)HG_N_HEDGES * 4, stream);
  hg_degree_kernel<<<(HG_N_INC + 255) / 256, 256, 0, stream>>>(
      inc_v, inc_e, dv, de, HG_N_INC);
  hg_invdeg_kernel<<<(HG_N_NODES + 255) / 256, 256, 0, stream>>>(
      dv, de, HG_N_NODES);

  // 2. bf16 conversions
  hg_cvt_bf16_kernel<<<(HG_N_NODES * 128 + 255) / 256, 256, 0, stream>>>(
      X, Xb, HG_N_NODES * 128);
  hg_cvt_bf16_kernel<<<(128 * 256 + 255) / 256, 256, 0, stream>>>(
      W1, W1b, 128 * 256);
  hg_cvt_bf16_kernel<<<(256 * 128 + 255) / 256, 256, 0, stream>>>(
      W2, W2b, 256 * 128);

  // 3. GEMM1: P = (X@W1 + b1) * dv_isqrt   [N,256]
  hg_gemm_wmma_kernel<<<dim3(HG_N_NODES / 32, 4), 256, 0, stream>>>(
      Xb, W1b, b1, dv, P, 128, 256);

  // 4. Ye: Q[e] += P[v]   (sorted-e segmented scan)
  hipMemsetAsync(Q, 0, (size_t)HG_N_HEDGES * 256 * 4, stream);
  hg_scatter_seg_kernel<<<dim3(HG_N_INC / 256, 64), 256, 0, stream>>>(
      P, Q, inc_v, inc_e, 256, HG_N_INC);

  // 5. Xo: R[v] += Q[e] * de_inv   (unsorted v -> atomic)
  hipMemsetAsync(R, 0, (size_t)HG_N_NODES * 256 * 4, stream);
  {
    long long tot = (long long)HG_N_INC * 64;
    hg_scatter_add_kernel<<<(unsigned)((tot + 255) / 256), 256, 0, stream>>>(
        Q, R, inc_e, inc_v, de, 64, HG_N_INC);
  }
  // 6. H = bf16(relu(R * dv_isqrt)) into P region
  bf16_t* Hb = (bf16_t*)P;
  hg_relu_scale_cvt_kernel<<<(HG_N_NODES * 256 + 255) / 256, 256, 0, stream>>>(
      R, dv, Hb, HG_N_NODES * 256, 8);

  // 7. GEMM2: Q = (H@W2 + b2) * dv_isqrt   [N,128]
  hg_gemm_wmma_kernel<<<dim3(HG_N_NODES / 32, 2), 256, 0, stream>>>(
      Hb, W2b, b2, dv, Q, 256, 128);

  // 8. Ye2: R[e] += Q[v]   (sorted-e segmented scan)
  hipMemsetAsync(R, 0, (size_t)HG_N_HEDGES * 128 * 4, stream);
  hg_scatter_seg_kernel<<<dim3(HG_N_INC / 256, 32), 256, 0, stream>>>(
      Q, R, inc_v, inc_e, 128, HG_N_INC);

  // 9. Z0: P[v] += R[e] * de_inv   (P region reused as f32 [N,128])
  float* Z0 = P;
  hipMemsetAsync(Z0, 0, (size_t)HG_N_NODES * 128 * 4, stream);
  {
    long long tot = (long long)HG_N_INC * 32;
    hg_scatter_add_kernel<<<(unsigned)((tot + 255) / 256), 256, 0, stream>>>(
        R, Z0, inc_e, inc_v, de, 32, HG_N_INC);
  }
  // 10. z = relu(Z0 * dv_isqrt) into Q   [N,128]
  float* z = Q;
  hg_relu_scale_kernel<<<(HG_N_NODES * 128 + 255) / 256, 256, 0, stream>>>(
      Z0, dv, z, HG_N_NODES * 128, 7);

  // 11. link pooling + fc + sigmoid
  hg_link_kernel<<<HG_N_LINKS / 8, 256, 0, stream>>>(
      z, link, fcW, fcb, out, HG_N_LINKS);
}
